// E57LearnedRadiusCell_45792941310465
// MI455X (gfx1250) — matrix operand order repaired and checked by reading, not compiled, not run on hardware
//
#include <hip/hip_runtime.h>
#include <cmath>

#define T_STEPS 512
#define B_DIM   64
#define D_DIM   1536
#define KB_CNT  (D_DIM / 32)            // 48 K-blocks of depth 32
#define DD      ((size_t)D_DIM * D_DIM)
#define HPLANE  ((size_t)4 * KB_CNT * 32 * 16)   // packed h A-frag plane: 4 mblocks
#define EPSV    1e-8f
#define MAX_RADIUS 0.999f

typedef __bf16 bf16_t;
typedef bf16_t v16bf __attribute__((ext_vector_type(16)));
typedef float  v8f   __attribute__((ext_vector_type(8)));

// ---------------- fragment helpers ----------------
// B-plane: fragment (ntile, kb) = 512 bf16; element el of lane l <-> B[k][n],
//   n = ntile*16 + (l&15),  k = kb*32 + ((l&16)?8:0) + (el&7) + ((el&8)?16:0)
__device__ __forceinline__ v16bf load_bfrag(const bf16_t* __restrict__ plane,
                                            int ntile, int kb, int lane) {
  return *(const v16bf*)(plane + (((size_t)ntile * KB_CNT + kb) << 9) + (lane << 4));
}

// Packed A-plane for h: fragment (mb, kb) = 512 bf16, same per-lane striping as the
// ISA 16x32 A layout (lane<16: K {0..7,16..23}; lane>=16: K {8..15,24..31}).
__device__ __forceinline__ v16bf load_afrag_packed(const bf16_t* __restrict__ plane,
                                                   int mb, int kb, int lane) {
  return *(const v16bf*)(plane + (((size_t)mb * KB_CNT + kb) << 9) + (lane << 4));
}

// Scatter one fp32 element of h into the split-bf16 packed A layout.
__device__ __forceinline__ void store_hpack(bf16_t* __restrict__ hhi, bf16_t* __restrict__ hlo,
                                            int m, int k, float v) {
  int mb = m >> 4, ml = m & 15, kb = k >> 5, kk = k & 31;
  int ln = ml + ((kk & 8) ? 16 : 0);
  int el = (kk & 7) + ((kk & 16) ? 8 : 0);
  size_t off = (((size_t)(mb * KB_CNT + kb) * 32 + ln) << 4) + el;
  bf16_t h = (bf16_t)v;
  hhi[off] = h;
  hlo[off] = (bf16_t)(v - (float)h);
}

// On-the-fly split-bf16 A fragment from an fp32 row-major matrix (for the x GEMM).
__device__ __forceinline__ void make_afrag(const float* __restrict__ arow,
                                           int kb, int lane, v16bf& hi, v16bf& lo) {
  const float* p = arow + kb * 32 + ((lane & 16) ? 8 : 0);
  float f[16];
  *(float4*)(f + 0)  = *(const float4*)(p + 0);
  *(float4*)(f + 4)  = *(const float4*)(p + 4);
  *(float4*)(f + 8)  = *(const float4*)(p + 16);
  *(float4*)(f + 12) = *(const float4*)(p + 20);
#pragma unroll
  for (int i = 0; i < 16; ++i) {
    bf16_t h = (bf16_t)f[i];
    hi[i] = h;
    lo[i] = (bf16_t)(f[i] - (float)h);
  }
}

#define WMMA_BF16(A, B, C) \
  __builtin_amdgcn_wmma_f32_16x16x32_bf16(false, (A), false, (B), (short)0, (C), false, false)

// ---------------- spectral-norm preamble (1 block) ----------------
__device__ __forceinline__ float block_sum(float v, float* red, int tid) {
  red[tid] = v; __syncthreads();
  for (int s = 128; s > 0; s >>= 1) {
    if (tid < s) red[tid] += red[tid + s];
    __syncthreads();
  }
  float r = red[0]; __syncthreads();
  return r;
}

__global__ void spectral_kernel(const float* __restrict__ Wh,
                                const float* __restrict__ logr,
                                const float* __restrict__ u_in,
                                float* __restrict__ scale_out) {
  __shared__ float su[D_DIM];
  __shared__ float sv[D_DIM];
  __shared__ float red[256];
  const int tid = threadIdx.x;
  for (int i = tid; i < D_DIM; i += 256) su[i] = u_in[i];
  __syncthreads();
  float nw_last = 0.f;
  for (int it = 0; it < 3; ++it) {
    float acc[6];
#pragma unroll
    for (int c = 0; c < 6; ++c) acc[c] = 0.f;
    for (int i = 0; i < D_DIM; ++i) {            // v = Wh^T @ u
      float ui = su[i];
      const float* row = Wh + (size_t)i * D_DIM;
#pragma unroll
      for (int c = 0; c < 6; ++c) acc[c] += row[tid + 256 * c] * ui;
    }
    float s = 0.f;
#pragma unroll
    for (int c = 0; c < 6; ++c) s += acc[c] * acc[c];
    float nv = sqrtf(block_sum(s, red, tid));
#pragma unroll
    for (int c = 0; c < 6; ++c) sv[tid + 256 * c] = acc[c] / (nv + EPSV);
    __syncthreads();
#pragma unroll
    for (int c = 0; c < 6; ++c) acc[c] = 0.f;
    for (int j = 0; j < D_DIM; ++j) {            // w = Wh @ v
      float vj = sv[j];
#pragma unroll
      for (int c = 0; c < 6; ++c) acc[c] += Wh[(size_t)(tid + 256 * c) * D_DIM + j] * vj;
    }
    s = 0.f;
#pragma unroll
    for (int c = 0; c < 6; ++c) s += acc[c] * acc[c];
    float nw = sqrtf(block_sum(s, red, tid));
    nw_last = nw;
#pragma unroll
    for (int c = 0; c < 6; ++c) su[tid + 256 * c] = acc[c] / (nw + EPSV);
    __syncthreads();
  }
  if (tid == 0) {
    float target = (1.f / (1.f + expf(-logr[0]))) * MAX_RADIUS;
    float sigma = nw_last * nw_last / (nw_last + EPSV);
    scale_out[0] = target / (sigma + EPSV);
  }
}

// ---------------- pack W_x and scaled W_h into bf16 hi/lo B-fragment planes ----------------
__global__ void pack_kernel(const float* __restrict__ Wx, const float* __restrict__ Wh,
                            const float* __restrict__ scale_p,
                            bf16_t* __restrict__ wxh, bf16_t* __restrict__ wxl,
                            bf16_t* __restrict__ whh, bf16_t* __restrict__ whl) {
  size_t idx = (size_t)blockIdx.x * 256 + threadIdx.x;   // 0 .. 2*DD
  int which = idx >= DD;
  size_t e = which ? idx - DD : idx;
  int frag = (int)(e >> 9), rem = (int)(e & 511);
  int lane = rem >> 4, el = rem & 15;
  int nt = frag / KB_CNT, kb = frag % KB_CNT;
  int n = nt * 16 + (lane & 15);
  int k = kb * 32 + ((lane & 16) ? 8 : 0) + (el & 7) + ((el & 8) << 1);
  float w = which ? Wh[(size_t)n * D_DIM + k] * scale_p[0]
                  : Wx[(size_t)n * D_DIM + k];
  bf16_t h = (bf16_t)w;
  bf16_t l = (bf16_t)(w - (float)h);
  if (which) { whh[e] = h; whl[e] = l; }
  else       { wxh[e] = h; wxl[e] = l; }
}

// ---------------- init: h_all[0] = h0, and pack h0 into A-frag buffer 0 ----------------
__global__ void init_h_kernel(const float* __restrict__ h0, float* __restrict__ hall,
                              bf16_t* __restrict__ hhi, bf16_t* __restrict__ hlo) {
  int i = blockIdx.x * 256 + threadIdx.x;
  float v = h0[i];
  hall[i] = v;
  store_hpack(hhi, hlo, i / D_DIM, i % D_DIM, v);
}

// ---------------- phase C: P = x @ W_x^T + b  (into outs region) ----------------
__global__ void xgemm_kernel(const float* __restrict__ x, const float* __restrict__ bias,
                             const bf16_t* __restrict__ wxh, const bf16_t* __restrict__ wxl,
                             float* __restrict__ P) {
  const int mbase = blockIdx.y * 64;
  const int nbase = blockIdx.x * 64;
  const int tid = threadIdx.x;
  const int wid = tid >> 5, lane = tid & 31;
  const int wm = wid >> 1, wn = wid & 1;
  const float* arow = x + (size_t)(mbase + wm * 16 + (lane & 15)) * D_DIM;
  const int nt0 = (nbase >> 4) + wn * 2;
  v8f a0h = {}, a0l = {}, a0x = {}, a1h = {}, a1l = {}, a1x = {};
#pragma unroll 2
  for (int kb = 0; kb < KB_CNT; ++kb) {
    v16bf ah, al;
    make_afrag(arow, kb, lane, ah, al);
    v16bf bh0 = load_bfrag(wxh, nt0,     kb, lane);
    v16bf bl0 = load_bfrag(wxl, nt0,     kb, lane);
    v16bf bh1 = load_bfrag(wxh, nt0 + 1, kb, lane);
    v16bf bl1 = load_bfrag(wxl, nt0 + 1, kb, lane);
    a0h = WMMA_BF16(ah, bh0, a0h);
    a0l = WMMA_BF16(al, bh0, a0l);
    a0x = WMMA_BF16(ah, bl0, a0x);
    a1h = WMMA_BF16(ah, bh1, a1h);
    a1l = WMMA_BF16(al, bh1, a1l);
    a1x = WMMA_BF16(ah, bl1, a1x);
  }
  v8f acc0 = a0h + a0l + a0x;
  v8f acc1 = a1h + a1l + a1x;
  const int mr = mbase + wm * 16 + ((lane & 16) ? 8 : 0);
  const int nc = nbase + wn * 32 + (lane & 15);
  const float b0 = bias[nc], b1 = bias[nc + 16];
#pragma unroll
  for (int r = 0; r < 8; ++r) {
    P[(size_t)(mr + r) * D_DIM + nc]      = acc0[r] + b0;
    P[(size_t)(mr + r) * D_DIM + nc + 16] = acc1[r] + b1;
  }
}

// ---------------- recurrent step t ----------------
// Reads pre-packed split-bf16 h fragments (no conversion on the WMMA path),
// writes h fp32 to h_all[t+1], out_t to outs[t], and packed h_new for step t+1.
__global__ void step_kernel(int t, float* __restrict__ outs, float* __restrict__ hall,
                            const float* __restrict__ z,
                            const bf16_t* __restrict__ whh, const bf16_t* __restrict__ whl,
                            const bf16_t* __restrict__ hin_hi, const bf16_t* __restrict__ hin_lo,
                            bf16_t* __restrict__ hout_hi, bf16_t* __restrict__ hout_lo) {
  const size_t base = (size_t)t * B_DIM * D_DIM;
  float* hnext = hall + base + (size_t)B_DIM * D_DIM;
  float* Pout = outs + base;              // holds P = x@Wx^T + b; overwritten with out_t
  const float* zt = z + base;
  const int tid = threadIdx.x;
  const int wid = tid >> 5, lane = tid & 31;
  const int wm = wid >> 1, wn = wid & 1;
  const int nbase = blockIdx.x * 64;
  const int nt0 = (nbase >> 4) + wn * 2;
  v8f a0h = {}, a0l = {}, a0x = {}, a1h = {}, a1l = {}, a1x = {};
#pragma unroll 2
  for (int kb = 0; kb < KB_CNT; ++kb) {
    v16bf ah = load_afrag_packed(hin_hi, wm, kb, lane);
    v16bf al = load_afrag_packed(hin_lo, wm, kb, lane);
    v16bf bh0 = load_bfrag(whh, nt0,     kb, lane);
    v16bf bl0 = load_bfrag(whl, nt0,     kb, lane);
    v16bf bh1 = load_bfrag(whh, nt0 + 1, kb, lane);
    v16bf bl1 = load_bfrag(whl, nt0 + 1, kb, lane);
    if (kb + 1 < KB_CNT)
      __builtin_prefetch(whh + (((size_t)nt0 * KB_CNT + kb + 1) << 9) + (lane << 4), 0, 1);
    a0h = WMMA_BF16(ah, bh0, a0h);
    a0l = WMMA_BF16(al, bh0, a0l);
    a0x = WMMA_BF16(ah, bl0, a0x);
    a1h = WMMA_BF16(ah, bh1, a1h);
    a1l = WMMA_BF16(al, bh1, a1l);
    a1x = WMMA_BF16(ah, bl1, a1x);
  }
  v8f acc0 = a0h + a0l + a0x;
  v8f acc1 = a1h + a1l + a1x;
  const int mr = wm * 16 + ((lane & 16) ? 8 : 0);
  const int nc = nbase + wn * 32 + (lane & 15);
#pragma unroll
  for (int r = 0; r < 8; ++r) {
    size_t i0 = (size_t)(mr + r) * D_DIM + nc;
    size_t i1 = i0 + 16;
    float h0v = tanhf(Pout[i0] + acc0[r]);
    float h1v = tanhf(Pout[i1] + acc1[r]);
    hnext[i0] = h0v;
    hnext[i1] = h1v;
    store_hpack(hout_hi, hout_lo, mr + r, nc, h0v);
    store_hpack(hout_hi, hout_lo, mr + r, nc + 16, h1v);
    float z0 = zt[i0], z1 = zt[i1];
    Pout[i0] = h0v * (z0 / (1.f + expf(-z0)));
    Pout[i1] = h1v * (z1 / (1.f + expf(-z1)));
  }
}

extern "C" void kernel_launch(void* const* d_in, const int* in_sizes, int n_in,
                              void* d_out, int out_size, void* d_ws, size_t ws_size,
                              hipStream_t stream) {
  const float* x    = (const float*)d_in[0];
  const float* z    = (const float*)d_in[1];
  const float* h0   = (const float*)d_in[2];
  const float* Wx   = (const float*)d_in[3];
  const float* Wh   = (const float*)d_in[4];
  const float* bias = (const float*)d_in[5];
  const float* logr = (const float*)d_in[6];
  const float* u    = (const float*)d_in[7];

  float* outs = (float*)d_out;                               // [T,B,D]
  float* hall = outs + (size_t)T_STEPS * B_DIM * D_DIM;      // [T+1,B,D]

  float*  scale = (float*)d_ws;                              // ws: ~20 MB
  bf16_t* whh = (bf16_t*)((char*)d_ws + 256);
  bf16_t* whl = whh + DD;
  bf16_t* wxh = whl + DD;
  bf16_t* wxl = wxh + DD;
  bf16_t* hbuf[2][2];                                        // [parity][hi/lo]
  hbuf[0][0] = wxl + DD;
  hbuf[0][1] = hbuf[0][0] + HPLANE;
  hbuf[1][0] = hbuf[0][1] + HPLANE;
  hbuf[1][1] = hbuf[1][0] + HPLANE;

  spectral_kernel<<<1, 256, 0, stream>>>(Wh, logr, u, scale);
  pack_kernel<<<(unsigned)((2 * DD) / 256), 256, 0, stream>>>(Wx, Wh, scale, wxh, wxl, whh, whl);
  init_h_kernel<<<(B_DIM * D_DIM) / 256, 256, 0, stream>>>(h0, hall, hbuf[0][0], hbuf[0][1]);
  xgemm_kernel<<<dim3(D_DIM / 64, (T_STEPS * B_DIM) / 64), 256, 0, stream>>>(x, bias, wxh, wxl, outs);
  for (int t = 0; t < T_STEPS; ++t) {
    int pi = t & 1, po = pi ^ 1;
    step_kernel<<<D_DIM / 64, 256, 0, stream>>>(t, outs, hall, z, whh, whl,
                                                hbuf[pi][0], hbuf[pi][1],
                                                hbuf[po][0], hbuf[po][1]);
  }
}
